// CopyDecoder_77970836291794
// MI455X (gfx1250) — matrix operand order repaired and checked by reading, not compiled, not run on hardware
//
#include <hip/hip_runtime.h>
#include <hip/hip_bf16.h>

// ---------------- problem constants ----------------
constexpr int B  = 64;
constexpr int H  = 512;
constexpr int E  = 256;
constexpr int V  = 50000;
constexpr int LT = 64;
constexpr int LN = 256;
constexpr int S  = LT + LN;          // 320
constexpr int ST = V + S;            // 50320
constexpr int T  = 16;
constexpr int KQ = H + LT + LN;      // 832 fused-K for the q projection
constexpr int KX = H + E;            // 768 GRU input width
constexpr int G3 = 3 * H;            // 1536

constexpr size_t SAMP_OFF = (size_t)B * T * ST;          // 51,527,680
constexpr size_t COV_OFF  = SAMP_OFF + (size_t)B * T;    // 51,528,704

typedef __attribute__((ext_vector_type(16))) _Float16 v16h;
typedef __attribute__((ext_vector_type(8)))  _Float16 v8h;
typedef __attribute__((ext_vector_type(8)))  float    v8f;

// ==================================================================
// WMMA GEMM:  C[M=64,N] = A[64,K]_f16 (row-major, row stride lda)
//                       x W[N,K]_f16 (row-major => already B^T layout)
//                       + bias[N]
// One wave32 per 16-wide N-tile; the wave owns ALL 4 M-tiles (M=64),
// so each weight (B) fragment is loaded once and reused by 4
// v_wmma_f32_16x16x32_f16 accumulators -> 4x less L2->WGP weight
// traffic and 4 WMMA per B-fragment load pair.
// ==================================================================
__global__ __launch_bounds__(256)
void wmma_gemm_f16(const _Float16* __restrict__ A, int lda,
                   const _Float16* __restrict__ W,
                   const float* __restrict__ bias,
                   float* __restrict__ C,
                   int N, int K)
{
    const int waveN = blockIdx.x * (blockDim.x >> 5) + (threadIdx.x >> 5);
    const int tN    = N >> 4;
    if (waveN >= tN) return;              // whole-wave exit: EXEC stays all-1 for WMMA
    const int lane  = threadIdx.x & 31;
    const int half  = lane >> 4;          // 0: lanes 0-15, 1: lanes 16-31
    const int l16   = lane & 15;

    const _Float16* Wrow = W + (size_t)(waveN * 16 + l16) * K;
    const _Float16* Ar0  = A + (size_t)( 0 + l16) * lda;   // M-tile 0
    const _Float16* Ar1  = A + (size_t)(16 + l16) * lda;   // M-tile 1
    const _Float16* Ar2  = A + (size_t)(32 + l16) * lda;   // M-tile 2
    const _Float16* Ar3  = A + (size_t)(48 + l16) * lda;   // M-tile 3

    v8f acc0 = {}, acc1 = {}, acc2 = {}, acc3 = {};
    for (int k0 = 0; k0 < K; k0 += 32) {
        __builtin_prefetch((const void*)(Wrow + k0 + 64), 0, 1); // global_prefetch_b8 on weight stream
        // B 32x16 fragment: column n = waveN*16+l16; lanes 0-15 K 0..7,16..23, lanes 16-31 K 8..15,24..31
        v8h b_lo = *(const v8h*)(Wrow + k0 + half * 8);
        v8h b_hi = *(const v8h*)(Wrow + k0 + 16 + half * 8);
        v16h bf = __builtin_shufflevector(b_lo, b_hi, 0,1,2,3,4,5,6,7,8,9,10,11,12,13,14,15);

        // A 16x32 fragments (ISA 7.12.2 16-bit A layout), one per M-tile
        {
            v8h a_lo = *(const v8h*)(Ar0 + k0 + half * 8);
            v8h a_hi = *(const v8h*)(Ar0 + k0 + 16 + half * 8);
            v16h af = __builtin_shufflevector(a_lo, a_hi, 0,1,2,3,4,5,6,7,8,9,10,11,12,13,14,15);
            acc0 = __builtin_amdgcn_wmma_f32_16x16x32_f16(false, af, false, bf, (short)0, acc0, false, false);
        }
        {
            v8h a_lo = *(const v8h*)(Ar1 + k0 + half * 8);
            v8h a_hi = *(const v8h*)(Ar1 + k0 + 16 + half * 8);
            v16h af = __builtin_shufflevector(a_lo, a_hi, 0,1,2,3,4,5,6,7,8,9,10,11,12,13,14,15);
            acc1 = __builtin_amdgcn_wmma_f32_16x16x32_f16(false, af, false, bf, (short)0, acc1, false, false);
        }
        {
            v8h a_lo = *(const v8h*)(Ar2 + k0 + half * 8);
            v8h a_hi = *(const v8h*)(Ar2 + k0 + 16 + half * 8);
            v16h af = __builtin_shufflevector(a_lo, a_hi, 0,1,2,3,4,5,6,7,8,9,10,11,12,13,14,15);
            acc2 = __builtin_amdgcn_wmma_f32_16x16x32_f16(false, af, false, bf, (short)0, acc2, false, false);
        }
        {
            v8h a_lo = *(const v8h*)(Ar3 + k0 + half * 8);
            v8h a_hi = *(const v8h*)(Ar3 + k0 + 16 + half * 8);
            v16h af = __builtin_shufflevector(a_lo, a_hi, 0,1,2,3,4,5,6,7,8,9,10,11,12,13,14,15);
            acc3 = __builtin_amdgcn_wmma_f32_16x16x32_f16(false, af, false, bf, (short)0, acc3, false, false);
        }
    }

    const int ncol = waveN * 16 + l16;
    const float bv = bias ? bias[ncol] : 0.0f;
    #pragma unroll
    for (int r = 0; r < 8; ++r) {         // C/D layout: VGPR r -> row r (+8 for upper half-lanes)
        const int mr = r + half * 8;
        C[(size_t)( 0 + mr) * N + ncol] = acc0[r] + bv;
        C[(size_t)(16 + mr) * N + ncol] = acc1[r] + bv;
        C[(size_t)(32 + mr) * N + ncol] = acc2[r] + bv;
        C[(size_t)(48 + mr) * N + ncol] = acc3[r] + bv;
    }
}

// ---------------- weight prep (once per launch) ----------------
__global__ void f32_to_f16_kernel(const float* __restrict__ src, _Float16* __restrict__ dst, int n)
{
    int i = blockIdx.x * blockDim.x + threadIdx.x;
    if (i < n) dst[i] = (_Float16)src[i];
}

__global__ void build_wq_kernel(const float* __restrict__ Wg, const float* __restrict__ Wct,
                                const float* __restrict__ Wcn,
                                const float* __restrict__ bg, const float* __restrict__ bct,
                                const float* __restrict__ bcn,
                                _Float16* __restrict__ Wq, float* __restrict__ bq)
{
    int h = blockIdx.x;                   // H rows
    for (int c = threadIdx.x; c < KQ; c += blockDim.x) {
        float v = (c < H)      ? Wg[h * H + c]
                : (c < H + LT) ? Wct[h * LT + (c - H)]
                               : Wcn[h * LN + (c - H - LT)];
        Wq[(size_t)h * KQ + c] = (_Float16)v;
    }
    if (threadIdx.x == 0) bq[h] = bg[h] + bct[h] + bcn[h];
}

// ---------------- per-launch state init ----------------
__global__ void init_dec0_kernel(float* __restrict__ out)
{
    size_t i = (size_t)blockIdx.x * blockDim.x + threadIdx.x;
    if (i >= (size_t)B * ST) return;
    int b = (int)(i / ST), v = (int)(i % ST);
    out[((size_t)b * T) * ST + v] = (v == 1) ? 0.0f : -INFINITY;   // log(sos one-hot)
}

__global__ void init_state_kernel(float* hid, float* covt, float* covn,
                                  float* cl, int* cur, float* out)
{
    int i = blockIdx.x * blockDim.x + threadIdx.x;
    if (i < B * H)  hid[i]  = 0.0f;
    if (i < B * LT) covt[i] = 0.0f;
    if (i < B * LN) covn[i] = 0.0f;
    if (i < B) { cur[i] = 1; out[SAMP_OFF + (size_t)i * T] = 1.0f; }
    if (i == 0) cl[0] = 0.0f;
}

// ---------------- per-step kernels ----------------
__global__ void embed_kernel(const float* __restrict__ emb, const int* __restrict__ cur,
                             float* __restrict__ embd, _Float16* __restrict__ x16)
{
    int b = blockIdx.x;
    int idx = cur[b];
    if (idx > V) idx = 3;
    if (idx < 0) idx = 0;
    if (idx > V - 1) idx = V - 1;
    for (int e = threadIdx.x; e < E; e += blockDim.x) {
        float v = emb[(size_t)idx * E + e];
        embd[b * E + e] = v;
        x16[b * KX + H + e] = (_Float16)v;   // x = [context | embedded]
    }
}

__global__ void build_aq_kernel(const float* __restrict__ hid, const float* __restrict__ covt,
                                const float* __restrict__ covn, _Float16* __restrict__ Aq)
{
    int b = blockIdx.x;
    for (int c = threadIdx.x; c < KQ; c += blockDim.x) {
        float v = (c < H)      ? hid[b * H + c]
                : (c < H + LT) ? covt[b * LT + (c - H)]
                               : covn[b * LN + (c - H - LT)];
        Aq[(size_t)b * KQ + c] = (_Float16)v;
    }
}

__global__ __launch_bounds__(256)
void attn_kernel(const float* __restrict__ enc, const float* __restrict__ q,
                 float* __restrict__ covt, float* __restrict__ covn,
                 float* __restrict__ attn, float* __restrict__ ctx,
                 float* __restrict__ ctxt, float* __restrict__ ctxn,
                 _Float16* __restrict__ x16, float* __restrict__ cl)
{
    int b = blockIdx.x, tid = threadIdx.x;
    __shared__ float qs[H];
    __shared__ float sc[S];
    __shared__ float red[256];

    for (int h = tid; h < H; h += 256) qs[h] = q[b * H + h];
    __syncthreads();

    for (int s = tid; s < S; s += 256) {
        const float* er = enc + ((size_t)b * S + s) * H;
        float acc = 0.0f;
        for (int h = 0; h < H; ++h) acc += er[h] * qs[h];
        sc[s] = acc;
    }
    __syncthreads();

    float m = sc[tid]; if (tid + 256 < S) m = fmaxf(m, sc[tid + 256]);
    red[tid] = m; __syncthreads();
    for (int s = 128; s > 0; s >>= 1) { if (tid < s) red[tid] = fmaxf(red[tid], red[tid + s]); __syncthreads(); }
    float mx = red[0]; __syncthreads();

    float sum = expf(sc[tid] - mx); if (tid + 256 < S) sum += expf(sc[tid + 256] - mx);
    red[tid] = sum; __syncthreads();
    for (int s = 128; s > 0; s >>= 1) { if (tid < s) red[tid] += red[tid + s]; __syncthreads(); }
    float inv = 1.0f / red[0]; __syncthreads();

    for (int s = tid; s < S; s += 256) sc[s] = expf(sc[s] - mx) * inv;
    __syncthreads();

    // write attention, coverage loss + coverage update
    float cpart = 0.0f;
    for (int s = tid; s < S; s += 256) {
        float a = sc[s];
        attn[b * S + s] = a;
        if (s < LT) { float c = covt[b * LT + s];      cpart += 0.5f * fminf(c, a); covt[b * LT + s]      = c + a; }
        else        { float c = covn[b * LN + s - LT]; cpart += 0.5f * fminf(c, a); covn[b * LN + s - LT] = c + a; }
    }
    red[tid] = cpart; __syncthreads();
    for (int s = 128; s > 0; s >>= 1) { if (tid < s) red[tid] += red[tid + s]; __syncthreads(); }
    if (tid == 0) atomicAdd(cl, red[0]);

    // contexts (full, tweet-part, news-part)
    for (int h = tid; h < H; h += 256) {
        float at = 0.0f, an = 0.0f;
        for (int s = 0;  s < LT; ++s) at += sc[s] * enc[((size_t)b * S + s) * H + h];
        for (int s = LT; s < S;  ++s) an += sc[s] * enc[((size_t)b * S + s) * H + h];
        float c = at + an;
        ctxt[b * H + h] = at;
        ctxn[b * H + h] = an;
        ctx [b * H + h] = c;
        x16[b * KX + h] = (_Float16)c;
    }
}

__global__ __launch_bounds__(256)
void gates_kernel(const float* __restrict__ ctx, const float* __restrict__ ctxt,
                  const float* __restrict__ ctxn, const float* __restrict__ hid,
                  const float* __restrict__ embd,
                  const float* Wgc, const float* bgc,
                  const float* Wst, const float* bst,
                  const float* Winp, const float* binp,
                  const float* Wct, const float* bct,
                  const float* Wcn, const float* bcn,
                  float* __restrict__ mix)
{
    int b = blockIdx.x, tid = threadIdx.x;
    float s1 = 0, s2 = 0, s3 = 0, s4 = 0, s5 = 0;
    for (int h = tid; h < H; h += 256) {
        s1 += ctx [b * H + h] * Wgc[h];
        s2 += hid [b * H + h] * Wst[h];
        s4 += ctxt[b * H + h] * Wct[h];
        s5 += ctxn[b * H + h] * Wcn[h];
    }
    for (int e = tid; e < E; e += 256) s3 += embd[b * E + e] * Winp[e];

    __shared__ float r1[256], r2[256], r3[256], r4[256], r5[256];
    r1[tid] = s1; r2[tid] = s2; r3[tid] = s3; r4[tid] = s4; r5[tid] = s5;
    __syncthreads();
    for (int s = 128; s > 0; s >>= 1) {
        if (tid < s) { r1[tid] += r1[tid+s]; r2[tid] += r2[tid+s]; r3[tid] += r3[tid+s];
                       r4[tid] += r4[tid+s]; r5[tid] += r5[tid+s]; }
        __syncthreads();
    }
    if (tid == 0) {
        float pg = 1.0f / (1.0f + expf(-(r1[0] + bgc[0] + r2[0] + bst[0] + r3[0] + binp[0])));
        float pt = 1.0f / (1.0f + expf(-(r4[0] + bct[0])));
        float pn = 1.0f / (1.0f + expf(-(r5[0] + bcn[0])));
        float mm = fmaxf(pg, fmaxf(pt, pn));
        float e0 = expf(pg - mm), e1 = expf(pt - mm), e2 = expf(pn - mm);
        float inv = 1.0f / (e0 + e1 + e2);
        mix[b * 3 + 0] = e0 * inv; mix[b * 3 + 1] = e1 * inv; mix[b * 3 + 2] = e2 * inv;
    }
}

__global__ void gru_combine_kernel(const float* __restrict__ gi, const float* __restrict__ gh,
                                   const float* __restrict__ hid,
                                   float* __restrict__ nh, _Float16* __restrict__ nh16)
{
    int i = blockIdx.x * blockDim.x + threadIdx.x;
    if (i >= B * H) return;
    int b = i >> 9, h = i & (H - 1);
    float ir = gi[b * G3 + h], iz = gi[b * G3 + H + h], in = gi[b * G3 + 2 * H + h];
    float hr = gh[b * G3 + h], hz = gh[b * G3 + H + h], hn = gh[b * G3 + 2 * H + h];
    float r = 1.0f / (1.0f + expf(-(ir + hr)));
    float z = 1.0f / (1.0f + expf(-(iz + hz)));
    float n = tanhf(in + r * hn);
    float v = (1.0f - z) * n + z * hid[i];
    nh[i] = v;
    nh16[i] = (_Float16)v;
}

__global__ __launch_bounds__(256)
void softmax_out_kernel(const float* __restrict__ logits, const float* __restrict__ mix,
                        float* __restrict__ probs)
{
    int b = blockIdx.x, tid = threadIdx.x;
    __shared__ float red[256];
    float m = -INFINITY;
    for (int v = tid; v < V; v += 256) m = fmaxf(m, logits[(size_t)b * V + v]);
    red[tid] = m; __syncthreads();
    for (int s = 128; s > 0; s >>= 1) { if (tid < s) red[tid] = fmaxf(red[tid], red[tid + s]); __syncthreads(); }
    float mx = red[0]; __syncthreads();
    float sum = 0.0f;
    for (int v = tid; v < V; v += 256) sum += expf(logits[(size_t)b * V + v] - mx);
    red[tid] = sum; __syncthreads();
    for (int s = 128; s > 0; s >>= 1) { if (tid < s) red[tid] += red[tid + s]; __syncthreads(); }
    float scale = mix[b * 3 + 0] / red[0];       // p_gen / Z
    for (int v = tid; v < ST; v += 256)
        probs[(size_t)b * ST + v] = (v < V) ? expf(logits[(size_t)b * V + v] - mx) * scale : 0.0f;
}

__global__ void mix_copy_kernel(float* __restrict__ probs, const float* __restrict__ attn,
                                const float* __restrict__ mix,
                                const int* __restrict__ tweets, const int* __restrict__ news)
{
    int b = blockIdx.x, tid = threadIdx.x;      // 320 threads
    if (tid < LT) {
        int tok = tweets[b * LT + tid];
        atomicAdd(&probs[(size_t)b * ST + tok], mix[b * 3 + 1] * attn[b * S + tid]);
    } else if (tid < S) {
        int tok = news[b * LN + (tid - LT)];
        atomicAdd(&probs[(size_t)b * ST + tok], mix[b * 3 + 2] * attn[b * S + tid]);
    }
}

__global__ __launch_bounds__(256)
void final_kernel(const float* __restrict__ probs, float* __restrict__ out,
                  int* __restrict__ cur, int t)
{
    int b = blockIdx.x, tid = threadIdx.x;
    __shared__ float bv[256];
    __shared__ int   bi[256];
    float best = -INFINITY; int bidx = ST;
    for (int v = tid; v < ST; v += 256) {
        float p = probs[(size_t)b * ST + v];
        out[((size_t)b * T + t) * ST + v] = logf(p);      // log(0) -> -inf, matches jnp.log
        if (p > best) { best = p; bidx = v; }             // v increases -> first max kept
    }
    bv[tid] = best; bi[tid] = bidx; __syncthreads();
    for (int s = 128; s > 0; s >>= 1) {
        if (tid < s) {
            if (bv[tid + s] > bv[tid] || (bv[tid + s] == bv[tid] && bi[tid + s] < bi[tid])) {
                bv[tid] = bv[tid + s]; bi[tid] = bi[tid + s];
            }
        }
        __syncthreads();
    }
    if (tid == 0) {
        cur[b] = bi[0];
        out[SAMP_OFF + (size_t)b * T + t] = (float)bi[0];
    }
}

__global__ void finalize_kernel(const float* __restrict__ cl, float* __restrict__ out)
{
    if (blockIdx.x == 0 && threadIdx.x == 0) out[COV_OFF] = cl[0] / (float)(T * B);
}

// ==================================================================
extern "C" void kernel_launch(void* const* d_in, const int* in_sizes, int n_in,
                              void* d_out, int out_size, void* d_ws, size_t ws_size,
                              hipStream_t stream)
{
    // ---- inputs (setup_inputs dict order, params leaves in insertion order) ----
    const float* enc     = (const float*)d_in[0];   // (B,S,H)
    const int*   tweets  = (const int*)  d_in[1];   // (B,LT)
    const int*   news    = (const int*)  d_in[2];   // (B,LN)
    // d_in[3] final_encoder_hidden (unused by reference scan), d_in[4..5] lengths (full)
    const float* emb     = (const float*)d_in[6];   // (V,E)
    const float* Wg      = (const float*)d_in[7];   const float* bg   = (const float*)d_in[8];
    const float* Wct     = (const float*)d_in[9];   const float* bct  = (const float*)d_in[10];
    const float* Wcn     = (const float*)d_in[11];  const float* bcn  = (const float*)d_in[12];
    const float* Wgc     = (const float*)d_in[13];  const float* bgc  = (const float*)d_in[14];
    const float* Wst     = (const float*)d_in[15];  const float* bst  = (const float*)d_in[16];
    const float* Winp    = (const float*)d_in[17];  const float* binp = (const float*)d_in[18];
    const float* Wout    = (const float*)d_in[19];  const float* bout = (const float*)d_in[20];
    const float* Wcpt    = (const float*)d_in[21];  const float* bcpt = (const float*)d_in[22];
    const float* Wcpn    = (const float*)d_in[23];  const float* bcpn = (const float*)d_in[24];
    const float* Wih     = (const float*)d_in[25];  // (3H, H+E)
    const float* Whh     = (const float*)d_in[26];  // (3H, H)
    const float* bih     = (const float*)d_in[27];
    const float* bhh     = (const float*)d_in[28];

    float* out = (float*)d_out;

    // ---- workspace carve ----
    char*  ws  = (char*)d_ws;
    size_t off = 0;
    auto alloc = [&](size_t bytes) -> void* {
        void* p = ws + off;
        off = (off + bytes + 255) & ~(size_t)255;
        return p;
    };
    _Float16* WQ16   = (_Float16*)alloc((size_t)H  * KQ * 2);
    _Float16* WIH16  = (_Float16*)alloc((size_t)G3 * KX * 2);
    _Float16* WHH16  = (_Float16*)alloc((size_t)G3 * H  * 2);
    _Float16* WOUT16 = (_Float16*)alloc((size_t)V  * H  * 2);
    float*    BQ     = (float*)   alloc((size_t)H * 4);
    _Float16* AQ16   = (_Float16*)alloc((size_t)B * KQ * 2);
    _Float16* X16    = (_Float16*)alloc((size_t)B * KX * 2);
    _Float16* NH16   = (_Float16*)alloc((size_t)B * H * 2);
    float*    HID0   = (float*)   alloc((size_t)B * H * 4);
    float*    HID1   = (float*)   alloc((size_t)B * H * 4);
    float*    COVT   = (float*)   alloc((size_t)B * LT * 4);
    float*    COVN   = (float*)   alloc((size_t)B * LN * 4);
    float*    EMBD   = (float*)   alloc((size_t)B * E * 4);
    float*    Q      = (float*)   alloc((size_t)B * H * 4);
    float*    ATTN   = (float*)   alloc((size_t)B * S * 4);
    float*    CTX    = (float*)   alloc((size_t)B * H * 4);
    float*    CTXT   = (float*)   alloc((size_t)B * H * 4);
    float*    CTXN   = (float*)   alloc((size_t)B * H * 4);
    float*    GI     = (float*)   alloc((size_t)B * G3 * 4);
    float*    GH     = (float*)   alloc((size_t)B * G3 * 4);
    float*    MIX    = (float*)   alloc((size_t)B * 3 * 4);
    float*    LOGITS = (float*)   alloc((size_t)B * V * 4);
    float*    PROBS  = (float*)   alloc((size_t)B * ST * 4);
    int*      IDX    = (int*)     alloc((size_t)B * 4);
    float*    CL     = (float*)   alloc(256);
    (void)ws_size; (void)in_sizes; (void)n_in; (void)out_size;

    // ---- once-per-launch: weight conversion to f16 (L2-resident working set) ----
    {
        int n;
        n = G3 * KX; f32_to_f16_kernel<<<(n + 255) / 256, 256, 0, stream>>>(Wih,  WIH16,  n);
        n = G3 * H;  f32_to_f16_kernel<<<(n + 255) / 256, 256, 0, stream>>>(Whh,  WHH16,  n);
        n = V  * H;  f32_to_f16_kernel<<<(n + 255) / 256, 256, 0, stream>>>(Wout, WOUT16, n);
        build_wq_kernel<<<H, 256, 0, stream>>>(Wg, Wct, Wcn, bg, bct, bcn, WQ16, BQ);
    }

    // ---- state init (deterministic every call) ----
    init_dec0_kernel<<<(int)(((size_t)B * ST + 255) / 256), 256, 0, stream>>>(out);
    init_state_kernel<<<(B * H + 255) / 256, 256, 0, stream>>>(HID0, COVT, COVN, CL, IDX, out);

    auto gemm = [&](const _Float16* A, int lda, const _Float16* W, const float* bias,
                    float* C, int N, int K) {
        int wavesN = N >> 4;                      // one wave per N-tile, all 4 M-tiles per wave
        int blocks = (wavesN + 7) / 8;            // 8 waves per 256-thread block
        wmma_gemm_f16<<<blocks, 256, 0, stream>>>(A, lda, W, bias, C, N, K);
    };

    float* hidA = HID0;   // hidden entering the step
    float* hidB = HID1;   // hidden produced by the step

    for (int t = 1; t < T; ++t) {
        embed_kernel<<<B, 256, 0, stream>>>(emb, IDX, EMBD, X16);
        build_aq_kernel<<<B, 256, 0, stream>>>(hidA, COVT, COVN, AQ16);

        // q = [h|covT|covN] x [Wg|Wct|Wcn]^T + (bg+bct+bcn)
        gemm(AQ16, KQ, WQ16, BQ, Q, H, KQ);

        attn_kernel<<<B, 256, 0, stream>>>(enc, Q, COVT, COVN, ATTN, CTX, CTXT, CTXN, X16, CL);
        gates_kernel<<<B, 256, 0, stream>>>(CTX, CTXT, CTXN, hidA, EMBD,
                                            Wgc, bgc, Wst, bst, Winp, binp,
                                            Wcpt, bcpt, Wcpn, bcpn, MIX);

        // GRU gates
        gemm(X16,  KX, WIH16, bih, GI, G3, KX);
        gemm(AQ16, KQ, WHH16, bhh, GH, G3, H);   // first H cols of AQ16 are f16(hidden)
        gru_combine_kernel<<<(B * H + 255) / 256, 256, 0, stream>>>(GI, GH, hidA, hidB, NH16);

        // vocab projection (dominant GEMM: 64x50000x512)
        gemm(NH16, H, WOUT16, bout, LOGITS, V, H);

        softmax_out_kernel<<<B, 256, 0, stream>>>(LOGITS, MIX, PROBS);
        mix_copy_kernel<<<B, S, 0, stream>>>(PROBS, ATTN, MIX, tweets, news);
        final_kernel<<<B, 256, 0, stream>>>(PROBS, out, IDX, t);

        float* tmp = hidA; hidA = hidB; hidB = tmp;
    }

    finalize_kernel<<<1, 32, 0, stream>>>(CL, out);
}
